// ChartUtil_5016521802320
// MI455X (gfx1250) — compile-verified, matrix operand order checked by reading, not run on hardware
//
#include <hip/hip_runtime.h>

typedef __attribute__((ext_vector_type(16))) _Float16 v16h;
typedef __attribute__((ext_vector_type(8)))  _Float16 v8h;
typedef __attribute__((ext_vector_type(8)))  float    v8f;

#define LEVEL  16
#define LPOS   48      // LENGTH - LEVEL
#define TOPK   2
#define SIZE   512
#define BATCH  32
#define NCELLS 904
#define NCOMBO 64      // LEVEL * TOPK * TOPK
#define NEGV   (-1e8f)

union V16U { v16h v; v8h h[2]; };

__device__ inline v8f zero8() {
  v8f z;
#pragma unroll
  for (int i = 0; i < 8; ++i) z[i] = 0.0f;
  return z;
}

// ---------------------------------------------------------------------------
// Repack a row-major f32 matrix (Krows x Ncols) into the per-lane B-fragment
// layout of v_wmma_f32_16x16x32_f16 (CDNA5 ISA 7.12.2, 16-bit operands):
//   dst[((ct*TK + ks)*32 + lane)*16 + j]  (f16)
//   lane -> column ct*16 + (lane&15); half = lane>>4
//   j<8  -> k = ks*32 + half*8 + j
//   j>=8 -> k = ks*32 + 16 + half*8 + (j-8)
// ---------------------------------------------------------------------------
__global__ void pack_b_f16(const float* __restrict__ src, _Float16* __restrict__ dst,
                           int Krows, int Ncols) {
  int e = blockIdx.x * blockDim.x + threadIdx.x;
  int total = Krows * Ncols;
  if (e >= total) return;
  int TK   = Krows >> 5;         // K-tiles of 32
  int j    = e & 15;
  int lane = (e >> 4) & 31;
  int ks   = (e >> 9) % TK;
  int ct   = e / (512 * TK);
  int half = lane >> 4;
  int col  = lane & 15;
  int k = (j < 8) ? (ks * 32 + half * 8 + j)
                  : (ks * 32 + 16 + half * 8 + (j - 8));
  dst[e] = (_Float16)src[k * Ncols + ct * 16 + col];
}

// ---------------------------------------------------------------------------
// Stage A: per (b,l): T = LH(32x512) @ mat (WMMA f16->f32), block-diag dots,
// + child scores, penalty mask, top-2, write scores/indices.
// ---------------------------------------------------------------------------
__global__ void __launch_bounds__(256)
score_topk_kernel(const float* __restrict__ chart_h,
                  const float* __restrict__ chart_s,
                  const int* __restrict__ l_index,
                  const int* __restrict__ r_index,
                  const _Float16* __restrict__ matB,
                  float* __restrict__ out,
                  int* __restrict__ ws_topk) {
  extern __shared__ char smem[];
  _Float16* Ah  = (_Float16*)smem;                       // 32 x 512 f16  (32 KB)
  _Float16* RHh = (_Float16*)(smem + 32768);             // 32 x 512 f16  (32 KB)
  float*    Tf  = (float*)(smem + 65536);                // 32 x 512 f32  (64 KB)
  float*    lsv = (float*)(smem + 131072);               // 32
  float*    rsv = lsv + 32;                              // 32
  float*    parts = rsv + 32;                            // 256
  float*    scores = parts + 256;                        // 64

  const int tid = threadIdx.x;
  const int p = blockIdx.x;
  const int b = p / LPOS, l = p % LPOS;

  // ---- gather left/right child rows, convert to f16 --------------------
  {
    int row = tid >> 3;                       // 0..31 = n*2 + k
    int sub = tid & 7;                        // 64 floats each
    int n = row >> 1, k = row & 1;
    long lcell = l_index[l * LEVEL + n];
    long rcell = r_index[l * LEVEL + n];
    const float* lsrc = chart_h + ((long)(k * BATCH + b)) * (long)(NCELLS * SIZE)
                        + lcell * SIZE + sub * 64;
    const float* rsrc = chart_h + ((long)(k * BATCH + b)) * (long)(NCELLS * SIZE)
                        + rcell * SIZE + sub * 64;
    _Float16* adst = Ah  + row * SIZE + sub * 64;
    _Float16* rdst = RHh + row * SIZE + sub * 64;
#pragma unroll 4
    for (int i = 0; i < 16; ++i) {
      float4 lv = ((const float4*)lsrc)[i];
      float4 rv = ((const float4*)rsrc)[i];
      adst[i * 4 + 0] = (_Float16)lv.x; adst[i * 4 + 1] = (_Float16)lv.y;
      adst[i * 4 + 2] = (_Float16)lv.z; adst[i * 4 + 3] = (_Float16)lv.w;
      rdst[i * 4 + 0] = (_Float16)rv.x; rdst[i * 4 + 1] = (_Float16)rv.y;
      rdst[i * 4 + 2] = (_Float16)rv.z; rdst[i * 4 + 3] = (_Float16)rv.w;
    }
  }
  if (tid < 32) {
    int n = tid >> 1, k = tid & 1;
    long lcell = l_index[l * LEVEL + n];
    long rcell = r_index[l * LEVEL + n];
    lsv[tid] = chart_s[(long)(k * BATCH + b) * NCELLS + lcell];
    rsv[tid] = chart_s[(long)(k * BATCH + b) * NCELLS + rcell];
  }
  __syncthreads();

  // ---- GEMM: T(32x512) = A(32x512) @ mat(512x512), f16 WMMA, f32 accum --
  {
    const int lane = tid & 31;
    const int w    = tid >> 5;                // 8 waves
    const int half = lane >> 4;
    const int arow = lane & 15;
    v8f acc[2][4];
#pragma unroll
    for (int r = 0; r < 2; ++r)
#pragma unroll
      for (int c = 0; c < 4; ++c) acc[r][c] = zero8();

    for (int ks = 0; ks < 16; ++ks) {
      const int c0 = ks * 32 + half * 8;
      V16U a0, a1;
      a0.h[0] = *(const v8h*)(Ah + arow * SIZE + c0);
      a0.h[1] = *(const v8h*)(Ah + arow * SIZE + c0 + 16);
      a1.h[0] = *(const v8h*)(Ah + (16 + arow) * SIZE + c0);
      a1.h[1] = *(const v8h*)(Ah + (16 + arow) * SIZE + c0 + 16);
#pragma unroll
      for (int c = 0; c < 4; ++c) {
        const int ct = w * 4 + c;
        const v16h bf = *(const v16h*)(matB + ((long)((ct * 16 + ks) * 32 + lane)) * 16);
        acc[0][c] = __builtin_amdgcn_wmma_f32_16x16x32_f16(
            false, a0.v, false, bf, (short)0, acc[0][c], false, false);
        acc[1][c] = __builtin_amdgcn_wmma_f32_16x16x32_f16(
            false, a1.v, false, bf, (short)0, acc[1][c], false, false);
      }
    }
    // C/D layout: VGPR i -> M=i (lanes 0-15) / M=i+8 (lanes 16-31), N=lane%16
    const int m0 = (lane < 16) ? 0 : 8;
    const int nn = lane & 15;
#pragma unroll
    for (int r = 0; r < 2; ++r)
#pragma unroll
      for (int c = 0; c < 4; ++c) {
        const int ct = w * 4 + c;
#pragma unroll
        for (int i = 0; i < 8; ++i)
          Tf[(r * 16 + m0 + i) * SIZE + ct * 16 + nn] = acc[r][c][i];
      }
  }
  __syncthreads();

  // ---- block-diagonal KxK dots: s[n,lk,rk] = T[n,lk] . RH[n,rk] ---------
  {
    int z = tid >> 2;                 // 0..63
    int part = tid & 3;               // 128 elems each
    int n = z >> 2, lk = (z >> 1) & 1, rk = z & 1;
    const float*    tr = Tf  + (n * 2 + lk) * SIZE + part * 128;
    const _Float16* rr = RHh + (n * 2 + rk) * SIZE + part * 128;
    float s = 0.0f;
#pragma unroll 8
    for (int i = 0; i < 128; ++i) s += tr[i] * (float)rr[i];
    parts[tid] = s;
  }
  __syncthreads();
  if (tid < NCOMBO) {
    int z = tid;
    int n = z >> 2, lk = (z >> 1) & 1, rk = z & 1;
    float s = parts[z * 4] + parts[z * 4 + 1] + parts[z * 4 + 2] + parts[z * 4 + 3]
            + lsv[n * 2 + lk] + rsv[n * 2 + rk];
    if (z == 2 || z == 3) s = NEGV;   // penalty mask: n==0, lk==1
    scores[z] = s;
  }
  __syncthreads();
  if (tid == 0) {
    int i1 = 0, i2 = 0;
    float s1 = -__builtin_inff(), s2 = -__builtin_inff();
    for (int z = 0; z < NCOMBO; ++z) {
      float s = scores[z];
      if (s > s1) { s2 = s1; i2 = i1; s1 = s; i1 = z; }
      else if (s > s2) { s2 = s; i2 = z; }
    }
    const long base = (long)(b * LPOS + l) * TOPK;
    const long offS = (long)BATCH * LPOS * TOPK * SIZE;   // 1,572,864
    float* outS  = out + offS;
    float* outN  = outS + (long)BATCH * LPOS * TOPK;
    float* outLK = outN + (long)BATCH * LPOS * TOPK;
    float* outRK = outLK + (long)BATCH * LPOS * TOPK;
    int   idx[2] = { i1, i2 };
    float sv[2]  = { s1, s2 };
    for (int kk = 0; kk < TOPK; ++kk) {
      int z = idx[kk];
      outS[base + kk]  = sv[kk];
      outN[base + kk]  = (float)(z >> 2);
      outLK[base + kk] = (float)((z >> 1) & 1);
      outRK[base + kk] = (float)(z & 1);
      ws_topk[base + kk] = z;
    }
  }
}

// ---------------------------------------------------------------------------
// Stage C: compose only the top-K winners.
// 16 selected rows per WG: H(16x512) = tanh(A(16x1024) @ Wc + bc), unit-norm.
// ---------------------------------------------------------------------------
__global__ void __launch_bounds__(256)
compose_kernel(const float* __restrict__ chart_h,
               const int* __restrict__ l_index,
               const int* __restrict__ r_index,
               const _Float16* __restrict__ WcB,
               const float* __restrict__ bc,
               const int* __restrict__ ws_topk,
               float* __restrict__ out) {
  extern __shared__ char smem[];
  _Float16* Ah    = (_Float16*)smem;            // 16 x 1024 f16 (32 KB)
  float*    Hf    = (float*)(smem + 32768);     // 16 x 512  f32 (32 KB)
  float*    parts = (float*)(smem + 65536);     // 256
  float*    rnorm = parts + 256;                // 16

  const int tid = threadIdx.x;
  const int R0  = blockIdx.x * 16;

  // ---- gather concat(lh_sel, rh_sel) rows -> f16 LDS --------------------
  {
    int i   = tid >> 4;               // row 0..15
    int sub = tid & 15;               // 64 floats each over 1024
    int grow = R0 + i;
    int b   = grow / (LPOS * TOPK);
    int rem = grow % (LPOS * TOPK);
    int l   = rem / TOPK;
    int z   = ws_topk[grow];
    int n = z >> 2, lk = (z >> 1) & 1, rk = z & 1;
    long cell; int k, dbase;
    if (sub < 8) { cell = l_index[l * LEVEL + n]; k = lk; dbase = sub * 64; }
    else         { cell = r_index[l * LEVEL + n]; k = rk; dbase = (sub - 8) * 64; }
    const float* src = chart_h + ((long)(k * BATCH + b)) * (long)(NCELLS * SIZE)
                       + cell * SIZE + dbase;
    _Float16* dst = Ah + i * 1024 + sub * 64;
#pragma unroll 4
    for (int q = 0; q < 16; ++q) {
      float4 v = ((const float4*)src)[q];
      dst[q * 4 + 0] = (_Float16)v.x; dst[q * 4 + 1] = (_Float16)v.y;
      dst[q * 4 + 2] = (_Float16)v.z; dst[q * 4 + 3] = (_Float16)v.w;
    }
  }
  __syncthreads();

  // ---- GEMM: H(16x512) = A(16x1024) @ Wc(1024x512) ----------------------
  {
    const int lane = tid & 31;
    const int w    = tid >> 5;
    const int half = lane >> 4;
    const int arow = lane & 15;
    v8f acc[4];
#pragma unroll
    for (int c = 0; c < 4; ++c) acc[c] = zero8();

    for (int ks = 0; ks < 32; ++ks) {
      const int c0 = ks * 32 + half * 8;
      V16U a;
      a.h[0] = *(const v8h*)(Ah + arow * 1024 + c0);
      a.h[1] = *(const v8h*)(Ah + arow * 1024 + c0 + 16);
#pragma unroll
      for (int c = 0; c < 4; ++c) {
        const int ct = w * 4 + c;
        const v16h bf = *(const v16h*)(WcB + ((long)((ct * 32 + ks) * 32 + lane)) * 16);
        acc[c] = __builtin_amdgcn_wmma_f32_16x16x32_f16(
            false, a.v, false, bf, (short)0, acc[c], false, false);
      }
    }
    const int m0 = (lane < 16) ? 0 : 8;
    const int nn = lane & 15;
#pragma unroll
    for (int c = 0; c < 4; ++c) {
      const int ct = w * 4 + c;
#pragma unroll
      for (int i = 0; i < 8; ++i)
        Hf[(m0 + i) * SIZE + ct * 16 + nn] = acc[c][i];
    }
  }
  __syncthreads();

  // ---- tanh(+bias), per-row sum of squares ------------------------------
  {
    int i = tid >> 4, sub = tid & 15;     // 32 elems each
    float ss = 0.0f;
#pragma unroll 4
    for (int q = 0; q < 32; ++q) {
      int d = sub * 32 + q;
      float v = tanhf(Hf[i * SIZE + d] + bc[d]);
      Hf[i * SIZE + d] = v;
      ss += v * v;
    }
    parts[tid] = ss;
  }
  __syncthreads();
  if (tid < 16) {
    float s = 0.0f;
    for (int q = 0; q < 16; ++q) s += parts[tid * 16 + q];
    rnorm[tid] = rsqrtf(s);
  }
  __syncthreads();
  {
    int i = tid >> 4, sub = tid & 15;
    int grow = R0 + i;
    float rn = rnorm[i];
    float* dst = out + (long)grow * SIZE + sub * 32;
#pragma unroll 4
    for (int q = 0; q < 32; ++q) dst[q] = Hf[i * SIZE + sub * 32 + q] * rn;
  }
}

// ---------------------------------------------------------------------------
extern "C" void kernel_launch(void* const* d_in, const int* in_sizes, int n_in,
                              void* d_out, int out_size, void* d_ws, size_t ws_size,
                              hipStream_t stream) {
  const float* chart_h = (const float*)d_in[0];
  const float* chart_s = (const float*)d_in[1];
  const int*   l_index = (const int*)d_in[2];
  const int*   r_index = (const int*)d_in[3];
  const float* mat     = (const float*)d_in[4];
  const float* Wc      = (const float*)d_in[5];
  const float* bc      = (const float*)d_in[6];
  float* out = (float*)d_out;

  char* ws = (char*)d_ws;
  _Float16* matB    = (_Float16*)ws;                                   // 512 KB
  _Float16* WcB     = (_Float16*)(ws + (size_t)512 * 512 * 2);         // 1 MB
  int*      ws_topk = (int*)(ws + (size_t)512 * 512 * 2 + (size_t)1024 * 512 * 2);

  // one-time operand repack (f32 -> f16 WMMA B-fragment layout)
  pack_b_f16<<<(512 * 512 + 255) / 256, 256, 0, stream>>>(mat, matB, 512, 512);
  pack_b_f16<<<(1024 * 512 + 255) / 256, 256, 0, stream>>>(Wc, WcB, 1024, 512);

  // stage A: scores + top-2
  size_t shA = 131072 + (32 + 32 + 256 + 64) * sizeof(float);
  score_topk_kernel<<<BATCH * LPOS, 256, shA, stream>>>(
      chart_h, chart_s, l_index, r_index, matB, out, ws_topk);

  // stage C: compose only winners
  size_t shC = 65536 + (256 + 16) * sizeof(float);
  compose_kernel<<<(BATCH * LPOS * TOPK) / 16, 256, shC, stream>>>(
      chart_h, l_index, r_index, WcB, bc, ws_topk, out);
}
